// CapsNet_81716047773878
// MI455X (gfx1250) — matrix operand order repaired and checked by reading, not compiled, not run on hardware
//
#include <hip/hip_runtime.h>
#include <hip/hip_bf16.h>

// CDNA5 wave32 WMMA types
typedef __attribute__((ext_vector_type(16))) _Float16 v16h;
typedef __attribute__((ext_vector_type(8)))  float    v8f;
typedef __attribute__((ext_vector_type(4)))  unsigned int v4u;

#define BATCH 256
#define NCLS  10
#define INCAPS 1152
#define INDIM  8
#define OUTDIM 16

// ---------------- workspace layout (bytes, 256-aligned) ----------------
constexpr size_t alignUp256(size_t x) { return (x + 255) & ~size_t(255); }
constexpr size_t OFF_X16  = 0;                                    // x  f16 [256,1,28,28]
constexpr size_t SZ_X16   = alignUp256((size_t)200704 * 2);
constexpr size_t OFF_W1   = OFF_X16 + SZ_X16;                     // conv1 w f16 [256,96] (81 padded)
constexpr size_t SZ_W1    = alignUp256((size_t)256 * 96 * 2);
constexpr size_t OFF_H16  = OFF_W1 + SZ_W1;                       // h  f16 [256,256,20,20]
constexpr size_t SZ_H16   = alignUp256((size_t)26214400 * 2);
constexpr size_t OFF_W2   = OFF_H16 + SZ_H16;                     // prim w f16 repacked [256, 81*256]
constexpr size_t SZ_W2    = alignUp256((size_t)5308416 * 2);
constexpr size_t OFF_P    = OFF_W2 + SZ_W2;                       // p  f32 [256,256,36]
constexpr size_t SZ_P     = alignUp256((size_t)2359296 * 4);
constexpr size_t OFF_U    = OFF_P + SZ_P;                         // u  f32 [256,1152,8]
constexpr size_t SZ_U     = alignUp256((size_t)2359296 * 4);
constexpr size_t OFF_VECS = OFF_U + SZ_U;                         // vecs f32 [256,10,16]
constexpr size_t SZ_VECS  = alignUp256((size_t)40960 * 4);
constexpr size_t OFF_DIN  = OFF_VECS + SZ_VECS;                   // dec_in [256,160]
constexpr size_t SZ_DIN   = alignUp256((size_t)40960 * 4);
constexpr size_t OFF_H1   = OFF_DIN + SZ_DIN;                     // [256,512]
constexpr size_t SZ_H1    = alignUp256((size_t)131072 * 4);
constexpr size_t OFF_H2   = OFF_H1 + SZ_H1;                       // [256,1024]

// ---------------- conversion / repack kernels ----------------
__global__ void k_f32_to_f16(const float* __restrict__ in, _Float16* __restrict__ out, int n) {
  int i = blockIdx.x * blockDim.x + threadIdx.x;
  if (i < n) out[i] = (_Float16)in[i];
}

__global__ void k_pad_conv1_w(const float* __restrict__ w, _Float16* __restrict__ out) {
  int i = blockIdx.x * blockDim.x + threadIdx.x;   // 256*96
  if (i >= 256 * 96) return;
  int oc = i / 96, k = i - oc * 96;
  out[i] = (k < 81) ? (_Float16)w[oc * 81 + k] : (_Float16)0.f;
}

// Repack prim_w [oc, ic, kh, kw] -> A[oc, k'] with k' = (kh*9+kw)*256 + ic.
// A 32-wide K-step is then one kernel position x 32 contiguous input channels.
__global__ void k_repack_prim_w(const float* __restrict__ w, _Float16* __restrict__ out) {
  int idx = blockIdx.x * blockDim.x + threadIdx.x;   // 256 * 20736
  if (idx >= 256 * 20736) return;
  int oc   = idx / 20736;
  int k    = idx - oc * 20736;
  int kpos = k >> 8;          // 0..80
  int ic   = k & 255;
  out[idx] = (_Float16)w[((size_t)oc * 256 + ic) * 81 + kpos];
}

// ---------------- conv1: implicit-GEMM WMMA with direct gather (K=81, x is L2-resident) ----------------
template<int IC, int IH, int IW, int OC, int OH, int OW, int STRIDE,
         int KREAL, int KPAD, bool RELU, bool STORE_F16>
__global__ void k_conv_wmma(const _Float16* __restrict__ A,     // [OC, KPAD] f16 weights
                            const _Float16* __restrict__ X,     // [BATCH, IC, IH, IW] f16
                            const float*    __restrict__ bias,  // [OC]
                            _Float16* __restrict__ outH,
                            float*    __restrict__ outF) {
  const int lane   = threadIdx.x & 31;
  const int waveId = blockIdx.x * (blockDim.x >> 5) + (threadIdx.x >> 5);
  constexpr int MT = OC / 16;
  const int mtile = (waveId % MT) * 16;
  const int ntile = (waveId / MT) * 16;

  const int mrow  = mtile + (lane & 15);
  const int abase = (lane < 16) ? 0 : 16;  // A 16x32 f16 layout
  const int ncol  = ntile + (lane & 15);
  const int kb2   = (lane < 16) ? 0 : 16;  // B 32x16 f16 layout

  const int b  = ncol / (OH * OW);
  const int rn = ncol - b * (OH * OW);
  const int oh = rn / OW;
  const int ow = rn - oh * OW;

  const _Float16* Arow = A + (size_t)mrow * KPAD;
  v8f acc = {};

  for (int k0 = 0; k0 < KPAD; k0 += 32) {
    union { v4u q[2]; v16h v; } af;
    af.q[0] = *(const v4u*)(Arow + k0 + abase);
    af.q[1] = *(const v4u*)(Arow + k0 + abase + 8);
    v16h bf;
#pragma unroll
    for (int e = 0; e < 16; ++e) {
      int k = k0 + kb2 + e;
      _Float16 val = (_Float16)0.f;
      if ((KPAD == KREAL) || (k < KREAL)) {
        int ic = k / 81;
        int kr = k - ic * 81;
        int kh = kr / 9;
        int kw = kr - kh * 9;
        int ih = oh * STRIDE + kh;
        int iw = ow * STRIDE + kw;
        val = X[(((size_t)b * IC + ic) * IH + ih) * IW + iw];
      }
      bf[e] = val;
    }
    acc = __builtin_amdgcn_wmma_f32_16x16x32_f16(false, af.v, false, bf,
                                                 (short)0, acc, false, false);
  }

  const int mbase = mtile + ((lane < 16) ? 0 : 8);
#pragma unroll
  for (int r = 0; r < 8; ++r) {
    int m = mbase + r;
    float val = acc[r] + bias[m];
    if (RELU) val = fmaxf(val, 0.f);
    size_t oidx = (((size_t)b * OC + m) * OH + oh) * OW + ow;
    if constexpr (STORE_F16) outH[oidx] = (_Float16)val;
    else                     outF[oidx] = val;
  }
}

// ---------------- primary caps conv (hotspot): LDS-staged blocked WMMA GEMM ----------------
// One block per batch image. 8 waves x (2 M-tiles x 3 N-tiles) cover oc=0..255, n=0..47 (36 valid).
// LDS holds h[b] chunk TRANSPOSED to [pixel][channel]; with the (kpos-major, ic-minor) weight
// repack, a lane's 16 B-fragment elements are 16 contiguous channels of one pixel ->
// each B fragment is just 2 x ds_load_b128 (16B aligned). Padded lanes (s>=36) only corrupt
// D columns that are never stored, so no value masking is needed (addresses are clamped).
__global__ void __launch_bounds__(256)
k_primary_wmma(const _Float16* __restrict__ A,   // [256, 20736] repacked (kpos*256 + ic)
               const _Float16* __restrict__ H,   // [256,256,20,20] f16
               const float*    __restrict__ bias,
               float*          __restrict__ P) { // [256,256,36] f32
  __shared__ __align__(64) _Float16 hs[400 * 32];   // 25.6 KB: [pixel 0..399][channel 0..31]

  const int tid  = threadIdx.x;
  const int lane = tid & 31;
  const int wave = tid >> 5;
  const int b    = blockIdx.x;

  const int l15   = lane & 15;
  const int abase = (lane < 16) ? 0 : 16;
  const int kb2   = (lane < 16) ? 0 : 16;

  const int mrow0 = (wave * 2) * 16 + l15;
  const _Float16* Arow0 = A + (size_t)mrow0 * 20736;
  const _Float16* Arow1 = Arow0 + (size_t)16 * 20736;

  // N decode per ntile (addresses clamped; no value mask needed)
  int ohv[3], owv[3];
#pragma unroll
  for (int nt = 0; nt < 3; ++nt) {
    int s  = nt * 16 + l15;
    int ss = (s < 36) ? s : 0;
    int oh = ss / 6;
    ohv[nt] = oh;
    owv[nt] = ss - oh * 6;
  }

  v8f acc[2][3];
#pragma unroll
  for (int mt = 0; mt < 2; ++mt)
#pragma unroll
    for (int nt = 0; nt < 3; ++nt) acc[mt][nt] = (v8f){};

  // staging role: lane = channel, wave index = pixel-group phase
  const int cch = tid & 31;
  const int pg0 = tid >> 5;

  for (int chunk = 0; chunk < 8; ++chunk) {
    const _Float16* src = H + ((size_t)b * 256 + chunk * 32) * 400;
    __builtin_prefetch((const void*)(src + 32 * 400), 0, 1);   // next chunk -> global_prefetch_b8
    __syncthreads();
    // transposing copy: hs[p*32 + c] = src[c*400 + p]
    for (int pg = pg0; pg < 50; pg += 8) {
      union { v4u q; _Float16 hv[8]; } t;
      t.q = *(const v4u*)(src + (size_t)cch * 400 + pg * 8);
#pragma unroll
      for (int j = 0; j < 8; ++j) hs[(pg * 8 + j) * 32 + cch] = t.hv[j];
    }
    __syncthreads();

    const int koff = chunk * 32;
    for (int kpos = 0; kpos < 81; ++kpos) {                    // uniform (SALU) loop
      const int kh = kpos / 9;
      const int kw = kpos - kh * 9;
      const int aoff = kpos * 256 + koff + abase;

      union { v4u q[2]; v16h v; } a0, a1;
      a0.q[0] = *(const v4u*)(Arow0 + aoff);
      a0.q[1] = *(const v4u*)(Arow0 + aoff + 8);
      a1.q[0] = *(const v4u*)(Arow1 + aoff);
      a1.q[1] = *(const v4u*)(Arow1 + aoff + 8);

#pragma unroll
      for (int nt = 0; nt < 3; ++nt) {
        const int hbase = (2 * ohv[nt] + kh) * 20 + 2 * owv[nt] + kw;   // <= 378 < 400
        const _Float16* bp = &hs[hbase * 32 + kb2];
        union { v4u q[2]; v16h v; } bf;
        bf.q[0] = *(const v4u*)bp;          // ds_load_b128
        bf.q[1] = *(const v4u*)(bp + 8);    // ds_load_b128
        acc[0][nt] = __builtin_amdgcn_wmma_f32_16x16x32_f16(false, a0.v, false, bf.v,
                                                            (short)0, acc[0][nt], false, false);
        acc[1][nt] = __builtin_amdgcn_wmma_f32_16x16x32_f16(false, a1.v, false, bf.v,
                                                            (short)0, acc[1][nt], false, false);
      }
    }
  }

  const int mb = (lane < 16) ? 0 : 8;
#pragma unroll
  for (int mt = 0; mt < 2; ++mt)
#pragma unroll
    for (int nt = 0; nt < 3; ++nt) {
      int s = nt * 16 + l15;
      if (s < 36) {
#pragma unroll
        for (int r = 0; r < 8; ++r) {
          int m = (wave * 2 + mt) * 16 + mb + r;
          P[((size_t)b * 256 + m) * 36 + s] = acc[mt][nt][r] + bias[m];
        }
      }
    }
}

// ---------------- squash primary capsules: p[b,oc,s] -> u[b,i,c] ----------------
__global__ void k_squash_u(const float* __restrict__ p, float* __restrict__ u) {
  int idx = blockIdx.x * blockDim.x + threadIdx.x;   // b*1152 + i
  if (idx >= BATCH * INCAPS) return;
  int b = idx / INCAPS, i = idx - b * INCAPS;
  int blk = i / 36, s = i - blk * 36;                // oc = c*32 + blk
  float t[8]; float sq = 0.f;
#pragma unroll
  for (int c = 0; c < 8; ++c) {
    t[c] = p[((size_t)b * 256 + c * 32 + blk) * 36 + s];
    sq += t[c] * t[c];
  }
  float scale = sqrtf(sq) / (1.f + sq);              // == (sq/(1+sq))/sqrt(sq)
#pragma unroll
  for (int c = 0; c < 8; ++c) u[(size_t)idx * 8 + c] = t[c] * scale;
}

// ---------------- dynamic routing: one block per (o,b) ----------------
__global__ void k_routing(const float* __restrict__ u, const float* __restrict__ route_w,
                          float* __restrict__ vecs) {
  __shared__ float u_s[INCAPS * INDIM];   // 36864 B
  __shared__ float logit_s[INCAPS];       // 4608 B
  __shared__ float red[256];
  __shared__ float s_s[OUTDIM];
  const int tid = threadIdx.x;
  const int o = blockIdx.x % NCLS;
  const int b = blockIdx.x / NCLS;

  for (int idx = tid; idx < INCAPS * INDIM; idx += 256)
    u_s[idx] = u[(size_t)b * INCAPS * INDIM + idx];
  for (int i = tid; i < INCAPS; i += 256) logit_s[i] = 0.f;
  __syncthreads();

  const float* rw = route_w + (size_t)o * INCAPS * INDIM * OUTDIM;
  float v[OUTDIM];

  for (int it = 0; it < 3; ++it) {
    float mloc = -3.0e38f;
    for (int i = tid; i < INCAPS; i += 256) mloc = fmaxf(mloc, logit_s[i]);
    red[tid] = mloc; __syncthreads();
    for (int st = 128; st > 0; st >>= 1) {
      if (tid < st) red[tid] = fmaxf(red[tid], red[tid + st]);
      __syncthreads();
    }
    float mx = red[0]; __syncthreads();
    float sl = 0.f;
    for (int i = tid; i < INCAPS; i += 256) sl += __expf(logit_s[i] - mx);
    red[tid] = sl; __syncthreads();
    for (int st = 128; st > 0; st >>= 1) {
      if (tid < st) red[tid] += red[tid + st];
      __syncthreads();
    }
    float den = red[0]; __syncthreads();

    if (tid < OUTDIM) s_s[tid] = 0.f;
    __syncthreads();

    float accd[OUTDIM];
#pragma unroll
    for (int d = 0; d < OUTDIM; ++d) accd[d] = 0.f;
    for (int i = tid; i < INCAPS; i += 256) {
      float pi = __expf(logit_s[i] - mx) / den;
      const float* rwi = rw + (size_t)i * 128;
      const float* ui  = &u_s[i * 8];
#pragma unroll
      for (int d = 0; d < OUTDIM; ++d) {
        float pr = 0.f;
#pragma unroll
        for (int c = 0; c < 8; ++c) pr += ui[c] * rwi[c * 16 + d];
        accd[d] += pi * pr;
      }
    }
#pragma unroll
    for (int d = 0; d < OUTDIM; ++d) atomicAdd(&s_s[d], accd[d]);   // ds_add_f32
    __syncthreads();

    float sq = 0.f;
#pragma unroll
    for (int d = 0; d < OUTDIM; ++d) sq += s_s[d] * s_s[d];
    float scale = sqrtf(sq) / (1.f + sq);
#pragma unroll
    for (int d = 0; d < OUTDIM; ++d) v[d] = s_s[d] * scale;
    __syncthreads();

    if (it != 2) {
      for (int i = tid; i < INCAPS; i += 256) {
        const float* rwi = rw + (size_t)i * 128;
        const float* ui  = &u_s[i * 8];
        float dv = 0.f;
#pragma unroll
        for (int d = 0; d < OUTDIM; ++d) {
          float pr = 0.f;
#pragma unroll
          for (int c = 0; c < 8; ++c) pr += ui[c] * rwi[c * 16 + d];
          dv += pr * v[d];
        }
        logit_s[i] += dv;
      }
      __syncthreads();
    }
  }
  if (tid < OUTDIM) vecs[((size_t)b * NCLS + o) * OUTDIM + tid] = v[tid];
}

// ---------------- class probs + masked decoder input ----------------
__global__ void k_classes(const float* __restrict__ vecs, float* __restrict__ cls_out,
                          float* __restrict__ dec_in) {
  int b = threadIdx.x;   // grid <<<1,256>>>
  float nrm[NCLS];
  for (int o = 0; o < NCLS; ++o) {
    const float* vv = vecs + ((size_t)b * NCLS + o) * OUTDIM;
    float s = 0.f;
    for (int d = 0; d < OUTDIM; ++d) s += vv[d] * vv[d];
    nrm[o] = sqrtf(s);
  }
  float mx = nrm[0];
  for (int o = 1; o < NCLS; ++o) mx = fmaxf(mx, nrm[o]);
  float e[NCLS]; float den = 0.f;
  for (int o = 0; o < NCLS; ++o) { e[o] = __expf(nrm[o] - mx); den += e[o]; }
  int am = 0; float best = -1.f;
  for (int o = 0; o < NCLS; ++o) {
    float c = e[o] / den;
    cls_out[b * NCLS + o] = c;
    if (c > best) { best = c; am = o; }        // first-max tie rule like argmax
  }
  for (int o = 0; o < NCLS; ++o) {
    const float* vv = vecs + ((size_t)b * NCLS + o) * OUTDIM;
    for (int d = 0; d < OUTDIM; ++d)
      dec_in[b * 160 + o * OUTDIM + d] = (o == am) ? vv[d] : 0.f;
  }
}

// ---------------- decoder GEMM + activation (small, f32 VALU) ----------------
template<int M, int K, int N, int ACT>   // ACT 0=relu, 1=sigmoid
__global__ void k_gemm(const float* __restrict__ A, const float* __restrict__ W,
                       const float* __restrict__ bias, float* __restrict__ out) {
  int idx = blockIdx.x * blockDim.x + threadIdx.x;
  if (idx >= M * N) return;
  int m = idx / N, n = idx - m * N;
  float acc = bias[n];
  const float* a = A + (size_t)m * K;
  const float* w = W + n;
  for (int k = 0; k < K; ++k) acc += a[k] * w[(size_t)k * N];
  if (ACT == 0) acc = fmaxf(acc, 0.f);
  else          acc = 1.f / (1.f + __expf(-acc));
  out[idx] = acc;
}

// ---------------- launch ----------------
extern "C" void kernel_launch(void* const* d_in, const int* in_sizes, int n_in,
                              void* d_out, int out_size, void* d_ws, size_t ws_size,
                              hipStream_t stream) {
  const float* x       = (const float*)d_in[0];
  const float* conv1_w = (const float*)d_in[1];
  const float* conv1_b = (const float*)d_in[2];
  const float* prim_w  = (const float*)d_in[3];
  const float* prim_b  = (const float*)d_in[4];
  const float* route_w = (const float*)d_in[5];
  const float* dec_w1  = (const float*)d_in[6];
  const float* dec_b1  = (const float*)d_in[7];
  const float* dec_w2  = (const float*)d_in[8];
  const float* dec_b2  = (const float*)d_in[9];
  const float* dec_w3  = (const float*)d_in[10];
  const float* dec_b3  = (const float*)d_in[11];
  float* out = (float*)d_out;
  char*  ws  = (char*)d_ws;

  _Float16* x16 = (_Float16*)(ws + OFF_X16);
  _Float16* w1f = (_Float16*)(ws + OFF_W1);
  _Float16* h16 = (_Float16*)(ws + OFF_H16);
  _Float16* w2f = (_Float16*)(ws + OFF_W2);
  float* p    = (float*)(ws + OFF_P);
  float* u    = (float*)(ws + OFF_U);
  float* vecs = (float*)(ws + OFF_VECS);
  float* din  = (float*)(ws + OFF_DIN);
  float* h1   = (float*)(ws + OFF_H1);
  float* h2   = (float*)(ws + OFF_H2);

  // precision prep: f16 multiply inputs, f32 WMMA accumulation
  k_f32_to_f16<<<(200704 + 255) / 256, 256, 0, stream>>>(x, x16, 200704);
  k_pad_conv1_w<<<(256 * 96 + 255) / 256, 256, 0, stream>>>(conv1_w, w1f);
  k_repack_prim_w<<<(5308416 + 255) / 256, 256, 0, stream>>>(prim_w, w2f);

  // conv1: M=256, N=102400, K=81 (pad 96); waves = 16*6400 -> 25600 blocks x 4 waves
  k_conv_wmma<1, 28, 28, 256, 20, 20, 1, 81, 96, true, true>
      <<<25600, 128, 0, stream>>>(w1f, x16, conv1_b, h16, nullptr);

  // primary caps conv hotspot: one block per image, transposed LDS staging,
  // B fragments = 2 x ds_load_b128, 6 WMMA per K-step
  k_primary_wmma<<<BATCH, 256, 0, stream>>>(w2f, h16, prim_b, p);

  k_squash_u<<<(BATCH * INCAPS + 255) / 256, 256, 0, stream>>>(p, u);
  k_routing<<<NCLS * BATCH, 256, 0, stream>>>(u, route_w, vecs);
  k_classes<<<1, 256, 0, stream>>>(vecs, out, din);

  k_gemm<256, 160, 512, 0><<<512, 256, 0, stream>>>(din, dec_w1, dec_b1, h1);
  k_gemm<256, 512, 1024, 0><<<1024, 256, 0, stream>>>(h1, dec_w2, dec_b2, h2);
  k_gemm<256, 1024, 784, 1><<<784, 256, 0, stream>>>(h2, dec_w3, dec_b3, out + BATCH * NCLS);
}
